// MultiPairsGeometricHead_81458349736066
// MI455X (gfx1250) — matrix operand order repaired and checked by reading, not compiled, not run on hardware
//
#include <hip/hip_runtime.h>

// ---------------------------------------------------------------------------
// MI455X (gfx1250) implementation.
// Heavy GEMMs -> V_WMMA_F32_16X16X32_BF16 (wave32), async global->LDS staging
// (GLOBAL_LOAD_ASYNC_TO_LDS_B128 + s_wait_asynccnt), bf16 data / fp32 accum.
// ---------------------------------------------------------------------------

typedef __attribute__((ext_vector_type(16))) __bf16 v16bf;
typedef __attribute__((ext_vector_type(8)))  __bf16 v8bf;
typedef __attribute__((ext_vector_type(8)))  float  v8f;

typedef unsigned short bfraw;   // raw bf16 bits in memory

__device__ __forceinline__ bfraw f2bf(float x) {
    union { float f; unsigned u; } v; v.f = x;
    unsigned r = (v.u + 0x7FFFu + ((v.u >> 16) & 1u)) >> 16;   // RNE
    return (bfraw)r;
}
__device__ __forceinline__ float bf2f(bfraw b) {
    union { unsigned u; float f; } v; v.u = ((unsigned)b) << 16;
    return v.f;
}

// Async 16B global -> LDS copy. LDS offset is the low 32 bits of the generic
// pointer (flat addressing truncates to the LDS aperture offset).
__device__ __forceinline__ void async_copy16(void* lds, const void* gaddr) {
    unsigned l = (unsigned)(size_t)lds;
    asm volatile("global_load_async_to_lds_b128 %0, %1, off"
                 :: "v"(l), "v"(gaddr) : "memory");
}
__device__ __forceinline__ void async_wait0() {
    asm volatile("s_wait_asynccnt 0x0" ::: "memory");
}

// ---------------------------------------------------------------------------
// Generic elementwise fp32 -> bf16 conversion
// ---------------------------------------------------------------------------
__global__ void cvt_kernel(const float* __restrict__ src, bfraw* __restrict__ dst, long n) {
    long i = (long)blockIdx.x * blockDim.x + threadIdx.x;
    long stride = (long)gridDim.x * blockDim.x;
    for (; i < n; i += stride) dst[i] = f2bf(src[i]);
}

// ---------------------------------------------------------------------------
// bf16 WMMA GEMM:  C[m,n] = sum_k A[m,k] * B[n,k]  (+ bias[n]) (+ relu)
// A: M x K row-major (lda). B: N x K row-major (ldb) == W[n,k] weight layout.
// Workgroup: 256 threads = 8 waves, tile 64(M) x 128(N), K-step 64.
// Each wave: 32x32 output = 2x2 WMMA tiles, 8 WMMAs per K iteration.
// Staging: GLOBAL_LOAD_ASYNC_TO_LDS_B128 (ASYNCcnt), no VGPR round-trip.
// ---------------------------------------------------------------------------
#define BLK_M 64
#define BLK_N 128
#define BLK_K 64
#define LDS_PITCH 72   // halves; 144B row pitch (16B multiple, conflict-breaking)

__global__ __launch_bounds__(256) void gemm_bf16_kernel(
    const bfraw* __restrict__ A, int lda,
    const bfraw* __restrict__ B, int ldb,
    const float* __restrict__ bias,
    bfraw* __restrict__ C, int ldc,
    int M, int N, int K, int relu)
{
    __shared__ bfraw As[BLK_M][LDS_PITCH];
    __shared__ bfraw Bs[BLK_N][LDS_PITCH];

    const int tid  = threadIdx.x;
    const int lane = tid & 31;
    const int wave = tid >> 5;
    const int wm   = wave >> 2;     // 0..1  -> 32 rows each
    const int wn   = wave & 3;      // 0..3  -> 32 cols each
    const int tileM = blockIdx.y * BLK_M;
    const int tileN = blockIdx.x * BLK_N;

    v8f acc[2][2];
    for (int mi = 0; mi < 2; ++mi)
        for (int ni = 0; ni < 2; ++ni)
            acc[mi][ni] = (v8f)(0.0f);

    const int rlo = lane & 15;
    const int lhi = lane >> 4;          // 0 or 1
    const int ahk = lhi * 8;            // A frag: K base {0,8} plus +16 chunk
    const int bhk = lhi * 8;            // B frag: K base {0,8} plus +16 chunk

    // Per-thread staging assignments (constant trip counts).
    const int arow0 = tid >> 3,            aseg0 = tid & 7;          // c = tid
    const int arow1 = (tid + 256) >> 3,    aseg1 = tid & 7;          // c = tid+256

    for (int k0 = 0; k0 < K; k0 += BLK_K) {
        // ---- A tile: 64 rows x 64 halves = 512 x 16B chunks, 2 per thread
        async_copy16(&As[arow0][aseg0 * 8],
                     A + (size_t)(tileM + arow0) * lda + k0 + aseg0 * 8);
        async_copy16(&As[arow1][aseg1 * 8],
                     A + (size_t)(tileM + arow1) * lda + k0 + aseg1 * 8);
        // ---- B tile: 128 rows x 64 halves = 1024 x 16B chunks, 4 per thread
        #pragma unroll
        for (int u = 0; u < 4; ++u) {
            const int c   = tid + u * 256;
            const int row = c >> 3, seg = c & 7;
            async_copy16(&Bs[row][seg * 8],
                         B + (size_t)(tileN + row) * ldb + k0 + seg * 8);
        }
        // ---- prefetch next K tile toward the WGP while async copies fly
        if (k0 + BLK_K < K) {
            __builtin_prefetch(A + (size_t)(tileM + (tid >> 2)) * lda + k0 + BLK_K, 0, 3);
            __builtin_prefetch(B + (size_t)(tileN + (tid >> 1)) * ldb + k0 + BLK_K, 0, 3);
        }
        async_wait0();
        __syncthreads();

        // ---- 2 WMMA K-steps per staging round: 8 v_wmma per iteration
        #pragma unroll
        for (int kk = 0; kk < BLK_K; kk += 32) {
            v16bf afrag[2], bfrag[2];
            #pragma unroll
            for (int mi = 0; mi < 2; ++mi) {
                const v8bf lo = *(const v8bf*)&As[wm * 32 + mi * 16 + rlo][kk + ahk];
                const v8bf hi = *(const v8bf*)&As[wm * 32 + mi * 16 + rlo][kk + ahk + 16];
                #pragma unroll
                for (int t = 0; t < 8; ++t) { afrag[mi][t] = lo[t]; afrag[mi][t + 8] = hi[t]; }
            }
            #pragma unroll
            for (int ni = 0; ni < 2; ++ni) {
                const v8bf lo = *(const v8bf*)&Bs[wn * 32 + ni * 16 + rlo][kk + bhk];
                const v8bf hi = *(const v8bf*)&Bs[wn * 32 + ni * 16 + rlo][kk + bhk + 16];
                #pragma unroll
                for (int t = 0; t < 8; ++t) { bfrag[ni][t] = lo[t]; bfrag[ni][t + 8] = hi[t]; }
            }
            #pragma unroll
            for (int mi = 0; mi < 2; ++mi)
                #pragma unroll
                for (int ni = 0; ni < 2; ++ni)
                    acc[mi][ni] = __builtin_amdgcn_wmma_f32_16x16x32_bf16(
                        false, afrag[mi], false, bfrag[ni],
                        (short)0, acc[mi][ni], false, false);
        }
        __syncthreads();
    }

    // ---- epilogue: C layout -> VGPR r, lanes 0-15: M=r,N=lane; lanes 16-31: M=8+r
    const int mhi = lhi * 8;
    for (int ni = 0; ni < 2; ++ni) {
        const int n = tileN + wn * 32 + ni * 16 + rlo;
        const float bv = bias ? bias[n] : 0.0f;
        for (int mi = 0; mi < 2; ++mi) {
            const int mbase = tileM + wm * 32 + mi * 16 + mhi;
            for (int r = 0; r < 8; ++r) {
                float v = acc[mi][ni][r] + bv;
                if (relu) v = fmaxf(v, 0.0f);
                C[(size_t)(mbase + r) * ldc + n] = f2bf(v);
            }
        }
    }
}

// ---------------------------------------------------------------------------
// neg mask + stable binary "argsort" (zeros first, index order) -> idx/enc_pad
// sigmoid(x) < 0.5  <=>  x < 0
// ---------------------------------------------------------------------------
__global__ __launch_bounds__(256) void negidx_kernel(
    const float* __restrict__ logits, const int* __restrict__ mask,
    float* __restrict__ neg, int* __restrict__ idx, int* __restrict__ encpad)
{
    const int b = blockIdx.x;
    const float* lg = logits + (long)b * 16384;
    for (int p = threadIdx.x; p < 16384; p += 256) {
        const int i = p >> 7, j = p & 127;
        const float pm = (mask[b * 128 + i] != 0 && mask[b * 128 + j] != 0) ? 1.0f : 0.0f;
        const float nb = (lg[p] < 0.0f) ? 1.0f : 0.0f;
        neg[(long)b * 16384 + p] = 1.0f - (1.0f - nb) * pm;
    }
    __syncthreads();
    if (threadIdx.x == 0) {
        int cnt = 0;
        for (int p = 0; p < 16384 && cnt < 96; ++p)
            if (neg[(long)b * 16384 + p] < 0.5f) { idx[b * 96 + cnt] = p; encpad[b * 96 + cnt] = 0; ++cnt; }
        for (int p = 0; p < 16384 && cnt < 96; ++p)
            if (neg[(long)b * 16384 + p] >= 0.5f) { idx[b * 96 + cnt] = p; encpad[b * 96 + cnt] = 1; ++cnt; }
        encpad[b * 96 + 0] = 0;
    }
}

// ---------------------------------------------------------------------------
// pos_pairs[b,r,k] = (A[b,i]+C[b,j]+b_rel)[k] * (1 - neg[p]),  p = idx[b,r]
// ---------------------------------------------------------------------------
__global__ void gather_pos_kernel(
    const bfraw* __restrict__ Abf, const bfraw* __restrict__ Cbf,
    const float* __restrict__ brel, const float* __restrict__ neg,
    const int* __restrict__ idx, bfraw* __restrict__ pos)
{
    const long total = (long)384 * 768;
    long e = (long)blockIdx.x * blockDim.x + threadIdx.x;
    for (; e < total; e += (long)gridDim.x * blockDim.x) {
        const int k = (int)(e % 768);
        const int r = (int)(e / 768);          // b*96 + slot
        const int b = r / 96;
        const int p = idx[r];
        const int i = p >> 7, j = p & 127;
        const float keep = (neg[(long)b * 16384 + p] > 0.5f) ? 0.0f : 1.0f;
        const float v = (bf2f(Abf[((long)(b * 128 + i)) * 768 + k]) +
                         bf2f(Cbf[((long)(b * 128 + j)) * 768 + k]) + brel[k]) * keep;
        pos[e] = f2bf(v);
    }
}

// ---------------------------------------------------------------------------
// X[b*16384+p, k] = A[b,i,k] + C[b,j,k] + b_rel[k]   (feat_pair, bf16)
// ---------------------------------------------------------------------------
__global__ void build_feat_kernel(
    const bfraw* __restrict__ Abf, const bfraw* __restrict__ Cbf,
    const float* __restrict__ brel, bfraw* __restrict__ X)
{
    const long total = (long)4 * 16384 * 768;
    long e = (long)blockIdx.x * blockDim.x + threadIdx.x;
    for (; e < total; e += (long)gridDim.x * blockDim.x) {
        const int  k    = (int)(e % 768);
        const long rowp = e / 768;
        const int  b    = (int)(rowp >> 14);
        const int  p    = (int)(rowp & 16383);
        const int  i = p >> 7, j = p & 127;
        const float v = bf2f(Abf[((long)(b * 128 + i)) * 768 + k]) +
                        bf2f(Cbf[((long)(b * 128 + j)) * 768 + k]) + brel[k];
        X[e] = f2bf(v);
    }
}

// ---------------------------------------------------------------------------
// Encoder self-attention: 96 tokens, 2 heads, dh=384. qkv row: [q|k|v] (2304).
// One block per (b,h); one thread per query token.
// ---------------------------------------------------------------------------
#define DH 384
__global__ __launch_bounds__(96) void enc_attn_kernel(
    const bfraw* __restrict__ qkv, const int* __restrict__ encpad,
    bfraw* __restrict__ ctx)
{
    __shared__ float sc[96][97];
    const int t = threadIdx.x;
    const int b = blockIdx.x >> 1;
    const int h = blockIdx.x & 1;
    const bfraw* qp = qkv + (long)(b * 96 + t) * 2304 + h * DH;
    const float scale = rsqrtf((float)DH);

    float mx = -1e30f;
    for (int s = 0; s < 96; ++s) {
        const bfraw* kp = qkv + (long)(b * 96 + s) * 2304 + 768 + h * DH;
        float acc = 0.0f;
        #pragma unroll 4
        for (int d = 0; d < DH; ++d) acc += bf2f(qp[d]) * bf2f(kp[d]);
        float sv = acc * scale;
        if (encpad[b * 96 + s]) sv = -1e9f;
        sc[t][s] = sv;
        mx = fmaxf(mx, sv);
    }
    float denom = 0.0f;
    for (int s = 0; s < 96; ++s) { float e = __expf(sc[t][s] - mx); sc[t][s] = e; denom += e; }
    const float inv = 1.0f / denom;
    for (int d = 0; d < DH; ++d) {
        float o = 0.0f;
        #pragma unroll 4
        for (int s = 0; s < 96; ++s)
            o += sc[t][s] * bf2f(qkv[(long)(b * 96 + s) * 2304 + 1536 + h * DH + d]);
        ctx[(long)(b * 96 + t) * 768 + h * DH + d] = f2bf(o * inv);
    }
}

// ---------------------------------------------------------------------------
// Decoder cross-attention: 65536 queries, 96 keys/batch, 2 heads, dh=384.
// Block = 128 threads: 64 queries x 2 heads. KV rows: [k(768) | v(768)].
// ---------------------------------------------------------------------------
__global__ __launch_bounds__(128) void dec_attn_kernel(
    const bfraw* __restrict__ Q, const bfraw* __restrict__ KV,
    const int* __restrict__ encpad, bfraw* __restrict__ ctx)
{
    __shared__ float sc[128][97];
    const int tid = threadIdx.x;
    const int ql  = tid >> 1;
    const int h   = tid & 1;
    const long qrow = (long)blockIdx.x * 64 + ql;
    const int  b    = (int)(qrow >> 14);
    const bfraw* qp = Q + qrow * 768 + h * DH;
    const float scale = rsqrtf((float)DH);

    float mx = -1e30f;
    for (int s = 0; s < 96; ++s) {
        const bfraw* kp = KV + (long)(b * 96 + s) * 1536 + h * DH;
        float acc = 0.0f;
        #pragma unroll 4
        for (int d = 0; d < DH; ++d) acc += bf2f(qp[d]) * bf2f(kp[d]);
        float sv = acc * scale;
        if (encpad[b * 96 + s]) sv = -1e9f;
        sc[tid][s] = sv;
        mx = fmaxf(mx, sv);
    }
    float denom = 0.0f;
    for (int s = 0; s < 96; ++s) { float e = __expf(sc[tid][s] - mx); sc[tid][s] = e; denom += e; }
    const float inv = 1.0f / denom;
    for (int d = 0; d < DH; ++d) {
        float o = 0.0f;
        #pragma unroll 4
        for (int s = 0; s < 96; ++s)
            o += sc[tid][s] * bf2f(KV[(long)(b * 96 + s) * 1536 + 768 + h * DH + d]);
        ctx[qrow * 768 + h * DH + d] = f2bf(o * inv);
    }
}

// ---------------------------------------------------------------------------
// out = LayerNorm(x + y) * g + b   (row of 768, eps=1e-5, population var)
// ---------------------------------------------------------------------------
__global__ __launch_bounds__(256) void add_ln_kernel(
    const bfraw* __restrict__ x, const bfraw* __restrict__ y,
    const float* __restrict__ g, const float* __restrict__ bta,
    bfraw* __restrict__ out)
{
    const long row = blockIdx.x;
    __shared__ float red0[8], red1[8];
    const int tid = threadIdx.x;
    float v[3], s = 0.0f, s2 = 0.0f;
    for (int i = 0; i < 3; ++i) {
        const int c = tid + i * 256;
        const float xv = bf2f(x[row * 768 + c]) + bf2f(y[row * 768 + c]);
        v[i] = xv; s += xv; s2 += xv * xv;
    }
    for (int off = 16; off; off >>= 1) { s += __shfl_down(s, off); s2 += __shfl_down(s2, off); }
    if ((tid & 31) == 0) { red0[tid >> 5] = s; red1[tid >> 5] = s2; }
    __syncthreads();
    if (tid == 0) {
        float a = 0.0f, q = 0.0f;
        for (int w = 0; w < 8; ++w) { a += red0[w]; q += red1[w]; }
        red0[0] = a; red1[0] = q;
    }
    __syncthreads();
    const float mean = red0[0] * (1.0f / 768.0f);
    const float var  = red1[0] * (1.0f / 768.0f) - mean * mean;
    const float inv  = rsqrtf(var + 1e-5f);
    for (int i = 0; i < 3; ++i) {
        const int c = tid + i * 256;
        out[row * 768 + c] = f2bf((v[i] - mean) * inv * g[c] + bta[c]);
    }
}

// ---------------------------------------------------------------------------
// out[r] = dot(t2[r,:], fc_w) + fc_b   -- one wave per row
// ---------------------------------------------------------------------------
__global__ __launch_bounds__(256) void fc_kernel(
    const bfraw* __restrict__ t2, const bfraw* __restrict__ w,
    const float* __restrict__ fb, float* __restrict__ out, long rows)
{
    const long r = (long)blockIdx.x * 8 + (threadIdx.x >> 5);
    if (r >= rows) return;
    const int lane = threadIdx.x & 31;
    float s = 0.0f;
    for (int d = lane; d < 768; d += 32) s += bf2f(t2[r * 768 + d]) * bf2f(w[d]);
    for (int off = 16; off; off >>= 1) s += __shfl_down(s, off);
    if (lane == 0) out[r] = s + fb[0];
}

// ---------------------------------------------------------------------------
// Host orchestration
// ---------------------------------------------------------------------------
extern "C" void kernel_launch(void* const* d_in, const int* in_sizes, int n_in,
                              void* d_out, int out_size, void* d_ws, size_t ws_size,
                              hipStream_t stream)
{
    const float* hs        = (const float*)d_in[0];
    const float* logits    = (const float*)d_in[1];
    const int*   bmask     = (const int*)  d_in[2];
    const float* W_rel     = (const float*)d_in[3];
    const float* b_rel     = (const float*)d_in[4];
    const float* enc_in_w  = (const float*)d_in[5];
    const float* enc_in_b  = (const float*)d_in[6];
    const float* enc_out_w = (const float*)d_in[7];
    const float* enc_out_b = (const float*)d_in[8];
    const float* enc_l1_w  = (const float*)d_in[9];
    const float* enc_l1_b  = (const float*)d_in[10];
    const float* enc_l2_w  = (const float*)d_in[11];
    const float* enc_l2_b  = (const float*)d_in[12];
    const float* enc_ln1_g = (const float*)d_in[13];
    const float* enc_ln1_b = (const float*)d_in[14];
    const float* enc_ln2_g = (const float*)d_in[15];
    const float* enc_ln2_b = (const float*)d_in[16];
    const float* dec_in_w  = (const float*)d_in[17];
    const float* dec_in_b  = (const float*)d_in[18];
    const float* dec_out_w = (const float*)d_in[19];
    const float* dec_out_b = (const float*)d_in[20];
    const float* dec_l1_w  = (const float*)d_in[21];
    const float* dec_l1_b  = (const float*)d_in[22];
    const float* dec_l2_w  = (const float*)d_in[23];
    const float* dec_l2_b  = (const float*)d_in[24];
    const float* dec_ln1_g = (const float*)d_in[25];
    const float* dec_ln1_b = (const float*)d_in[26];
    const float* dec_ln2_g = (const float*)d_in[27];
    const float* dec_ln2_b = (const float*)d_in[28];
    const float* fc_w      = (const float*)d_in[29];
    const float* fc_b      = (const float*)d_in[30];
    (void)in_sizes; (void)n_in; (void)out_size; (void)ws_size;

    // bump allocator over d_ws (256B aligned)
    char* ws = (char*)d_ws;
    size_t off = 0;
    auto alloc = [&](size_t bytes) -> void* {
        off = (off + 255) & ~(size_t)255;
        void* p = ws + off;
        off += bytes;
        return p;
    };

    const long RBN  = 512;          // B*N
    const long RENC = 384;          // B*96
    const long RPAIR = 65536;       // B*N*N

    bfraw* hs_bf     = (bfraw*)alloc(RBN * 768 * 2);
    bfraw* wrel_bf   = (bfraw*)alloc((size_t)768 * 1536 * 2);
    bfraw* A_bf      = (bfraw*)alloc(RBN * 768 * 2);
    bfraw* C_bf      = (bfraw*)alloc(RBN * 768 * 2);
    bfraw* encin_bf  = (bfraw*)alloc((size_t)2304 * 768 * 2);
    bfraw* encout_bf = (bfraw*)alloc((size_t)768 * 768 * 2);
    bfraw* encl1_bf  = (bfraw*)alloc((size_t)768 * 768 * 2);
    bfraw* encl2_bf  = (bfraw*)alloc((size_t)768 * 768 * 2);
    bfraw* decin_bf  = (bfraw*)alloc((size_t)2304 * 768 * 2);
    bfraw* decout_bf = (bfraw*)alloc((size_t)768 * 768 * 2);
    bfraw* decl1_bf  = (bfraw*)alloc((size_t)768 * 768 * 2);
    bfraw* decl2_bf  = (bfraw*)alloc((size_t)768 * 768 * 2);
    bfraw* fcw_bf    = (bfraw*)alloc((size_t)768 * 2);
    float* negbuf    = (float*)alloc((size_t)4 * 16384 * 4);
    int*   idxbuf    = (int*)  alloc((size_t)4 * 96 * 4);
    int*   padbuf    = (int*)  alloc((size_t)4 * 96 * 4);
    bfraw* pos_bf    = (bfraw*)alloc(RENC * 768 * 2);
    bfraw* qkv_bf    = (bfraw*)alloc(RENC * 2304 * 2);
    bfraw* ctx_e     = (bfraw*)alloc(RENC * 768 * 2);
    bfraw* mha_e     = (bfraw*)alloc(RENC * 768 * 2);
    bfraw* x1_bf     = (bfraw*)alloc(RENC * 768 * 2);
    bfraw* f1_bf     = (bfraw*)alloc(RENC * 768 * 2);
    bfraw* f2_bf     = (bfraw*)alloc(RENC * 768 * 2);
    bfraw* pat_bf    = (bfraw*)alloc(RENC * 768 * 2);
    bfraw* kv_bf     = (bfraw*)alloc(RENC * 1536 * 2);
    bfraw* Xbuf      = (bfraw*)alloc((size_t)RPAIR * 768 * 2);   // ~100 MiB
    bfraw* Buf1      = (bfraw*)alloc((size_t)RPAIR * 768 * 2);
    bfraw* Buf2      = (bfraw*)alloc((size_t)RPAIR * 768 * 2);

    auto cvt = [&](const float* src, bfraw* dst, long n) {
        long blocks = (n + 255) / 256;
        if (blocks > 8192) blocks = 8192;
        cvt_kernel<<<(int)blocks, 256, 0, stream>>>(src, dst, n);
    };
    auto gemm = [&](const bfraw* Ap, int lda, const bfraw* Bp, int ldb,
                    const float* bias, bfraw* Cp, int ldc,
                    int M, int N, int K, int relu) {
        dim3 grid(N / BLK_N, M / BLK_M);
        gemm_bf16_kernel<<<grid, 256, 0, stream>>>(Ap, lda, Bp, ldb, bias, Cp, ldc, M, N, K, relu);
    };

    // -- bf16 conversions (weights + hidden states)
    cvt(hs,        hs_bf,     RBN * 768);
    cvt(W_rel,     wrel_bf,   (long)768 * 1536);
    cvt(enc_in_w,  encin_bf,  (long)2304 * 768);
    cvt(enc_out_w, encout_bf, (long)768 * 768);
    cvt(enc_l1_w,  encl1_bf,  (long)768 * 768);
    cvt(enc_l2_w,  encl2_bf,  (long)768 * 768);
    cvt(dec_in_w,  decin_bf,  (long)2304 * 768);
    cvt(dec_out_w, decout_bf, (long)768 * 768);
    cvt(dec_l1_w,  decl1_bf,  (long)768 * 768);
    cvt(dec_l2_w,  decl2_bf,  (long)768 * 768);
    cvt(fc_w,      fcw_bf,    768);

    // -- A = hs @ Wa^T ; C = hs @ Wb^T   (Wa = W_rel[:, :H], Wb = W_rel[:, H:])
    gemm(hs_bf, 768, wrel_bf,        1536, nullptr, A_bf, 768, 512, 768, 768, 0);
    gemm(hs_bf, 768, wrel_bf + 768,  1536, nullptr, C_bf, 768, 512, 768, 768, 0);

    // -- neg mask / pair selection / encoder padding
    negidx_kernel<<<4, 256, 0, stream>>>(logits, bmask, negbuf, idxbuf, padbuf);
    gather_pos_kernel<<<1152, 256, 0, stream>>>(A_bf, C_bf, b_rel, negbuf, idxbuf, pos_bf);

    // -- encoder
    gemm(pos_bf, 768, encin_bf, 768, enc_in_b, qkv_bf, 2304, 384, 2304, 768, 0);
    enc_attn_kernel<<<8, 96, 0, stream>>>(qkv_bf, padbuf, ctx_e);
    gemm(ctx_e, 768, encout_bf, 768, enc_out_b, mha_e, 768, 384, 768, 768, 0);
    add_ln_kernel<<<384, 256, 0, stream>>>(pos_bf, mha_e, enc_ln1_g, enc_ln1_b, x1_bf);
    gemm(x1_bf, 768, encl1_bf, 768, enc_l1_b, f1_bf, 768, 384, 768, 768, 1);
    gemm(f1_bf, 768, encl2_bf, 768, enc_l2_b, f2_bf, 768, 384, 768, 768, 0);
    add_ln_kernel<<<384, 256, 0, stream>>>(x1_bf, f2_bf, enc_ln2_g, enc_ln2_b, pat_bf);

    // -- decoder K/V from patterns (rows 768..2303 of dec_in_w)
    gemm(pat_bf, 768, decin_bf + (size_t)768 * 768, 768, dec_in_b + 768,
         kv_bf, 1536, 384, 1536, 768, 0);

    // -- feat_pair (bf16) and decoder pipeline over 65536 tokens
    build_feat_kernel<<<4096, 256, 0, stream>>>(A_bf, C_bf, b_rel, Xbuf);
    gemm(Xbuf, 768, decin_bf, 768, dec_in_b, Buf1, 768, 65536, 768, 768, 0);      // Q
    dec_attn_kernel<<<1024, 128, 0, stream>>>(Buf1, kv_bf, padbuf, Buf2);         // ctx
    gemm(Buf2, 768, decout_bf, 768, dec_out_b, Buf1, 768, 65536, 768, 768, 0);    // out proj
    add_ln_kernel<<<65536, 256, 0, stream>>>(Xbuf, Buf1, dec_ln1_g, dec_ln1_b, Buf2);  // t1
    gemm(Buf2, 768, decl1_bf, 768, dec_l1_b, Buf1, 768, 65536, 768, 768, 1);      // relu FF1
    gemm(Buf1, 768, decl2_bf, 768, dec_l2_b, Xbuf, 768, 65536, 768, 768, 0);      // FF2
    add_ln_kernel<<<65536, 256, 0, stream>>>(Buf2, Xbuf, dec_ln2_g, dec_ln2_b, Buf1);  // t2

    // -- final projection to logits
    fc_kernel<<<8192, 256, 0, stream>>>(Buf1, fcw_bf, fc_b, (float*)d_out, RPAIR);
}
